// MultiHeadAttention_39238821216887
// MI455X (gfx1250) — matrix operand order repaired and checked by reading, not compiled, not run on hardware
//
#include <hip/hip_runtime.h>
#include <hip/hip_bf16.h>

// MHA: B=4, T=2048, D=1024, H=16, Hd=64.
// bf16 WMMA + flash-attention + async global->LDS double-buffered staging.
#define BATCH 4
#define T     2048
#define D     1024
#define H     16
#define HD    64
#define BT    (BATCH * T)   // 8192

typedef __attribute__((ext_vector_type(16))) __bf16 v16bf;
typedef __attribute__((ext_vector_type(8)))  __bf16 v8bf;
typedef __attribute__((ext_vector_type(8)))  float  v8f;
typedef __attribute__((ext_vector_type(4)))  int    v4i;

typedef __attribute__((address_space(1))) v4i* gptr_v4i;
typedef __attribute__((address_space(3))) v4i* lptr_v4i;

// Scheduling pins: 0x20 = VMEM read, 0x100 = DS read, 0x200 = DS write,
// 0x8 = MFMA/WMMA.  SGB(m,s): next s instructions of class m.
#define SGB(m, s) __builtin_amdgcn_sched_group_barrier((m), (s), 0)
#define PIPE1 SGB(0x100, 2); SGB(0x8, 1);
#define PIPE5 PIPE1 PIPE1 PIPE1 PIPE1 PIPE1

static __device__ __forceinline__ unsigned short f2bf(float f) {
  unsigned u = __builtin_bit_cast(unsigned, f);
  u += 0x7FFFu + ((u >> 16) & 1u);          // round-to-nearest-even
  return (unsigned short)(u >> 16);
}

static __device__ __forceinline__ v8bf ld8(const unsigned short* p) {
  return *reinterpret_cast<const v8bf*>(p);
}

static __device__ __forceinline__ v16bf cat16(v8bf lo, v8bf hi) {
  v16bf r;
#pragma unroll
  for (int i = 0; i < 8; ++i) { r[i] = lo[i]; r[i + 8] = hi[i]; }
  return r;
}

static __device__ __forceinline__ v8f vzero() {
  v8f z;
#pragma unroll
  for (int i = 0; i < 8; ++i) z[i] = 0.f;
  return z;
}

static __device__ __forceinline__ v8f wmma_bf16(v16bf a, v16bf b, v8f c) {
  return __builtin_amdgcn_wmma_f32_16x16x32_bf16(false, a, false, b, (short)0, c,
                                                 false, false);
}

// A-operand (16x32, 16-bit): lane<16 holds K {0..7,16..23}, lane>=16 holds K {8..15,24..31}.
static __device__ __forceinline__ v16bf loadA(const unsigned short* base, int kc, int lane) {
  const int klo = (lane < 16) ? 0 : 8;
  return cat16(ld8(base + kc + klo), ld8(base + kc + klo + 16));
}

// B-operand (32x16, 16-bit): lane<16 holds K {0..7,16..23}, lane>=16 holds K {8..15,24..31}.
static __device__ __forceinline__ v16bf loadB(const unsigned short* base, int kc, int lane) {
  const int klo = (lane < 16) ? 0 : 8;
  return cat16(ld8(base + kc + klo), ld8(base + kc + klo + 16));
}

// Async copy: 16B per lane, global -> LDS (gfx1250 GLOBAL_LOAD_ASYNC_TO_LDS_B128).
#if defined(__has_builtin)
#if __has_builtin(__builtin_amdgcn_global_load_async_to_lds_b128)
#define ASYNC_LDS_BUILTIN 1
#endif
#endif

static __device__ __forceinline__ void async_g2l_b128(unsigned short* lds,
                                                      const unsigned short* g) {
#ifdef ASYNC_LDS_BUILTIN
  __builtin_amdgcn_global_load_async_to_lds_b128(
      (gptr_v4i)(v4i*)(unsigned short*)g, (lptr_v4i)(v4i*)lds, 0, 0);
#else
  unsigned lds_addr =
      (unsigned)(size_t)(__attribute__((address_space(3))) unsigned short*)lds;
  asm volatile("global_load_async_to_lds_b128 %0, %1, off"
               :: "v"(lds_addr), "v"((unsigned long long)(size_t)g)
               : "memory");
#endif
}
static __device__ __forceinline__ void wait_async0() {
  asm volatile("s_wait_asynccnt 0x0" ::: "memory");
}

// ---------------------------------------------------------------- converts
__global__ __launch_bounds__(256) void cvt_bf16_kernel(const float* __restrict__ in,
                                                       unsigned short* __restrict__ out,
                                                       int n) {
  int i = (blockIdx.x * 256 + threadIdx.x) * 4;
  if (i + 3 < n) {
    float4 f = *reinterpret_cast<const float4*>(in + i);
    out[i + 0] = f2bf(f.x);
    out[i + 1] = f2bf(f.y);
    out[i + 2] = f2bf(f.z);
    out[i + 3] = f2bf(f.w);
  }
}

// W [K=1024][N=1024] f32 -> WT [N][K] bf16 so B-operand rows are contiguous.
__global__ __launch_bounds__(256) void transpose_bf16_kernel(const float* __restrict__ W,
                                                             unsigned short* __restrict__ WT) {
  int i = blockIdx.x * 256 + threadIdx.x;
  int n = i >> 10, k = i & 1023;
  WT[(size_t)n * D + k] = f2bf(W[(size_t)k * D + n]);
}

// ---------------------------------------------------------------- fused QKV projection
__global__ __launch_bounds__(256, 1)
void qkv_kernel(const unsigned short* __restrict__ xb,
                const unsigned short* __restrict__ WqT,
                const unsigned short* __restrict__ WkT,
                const unsigned short* __restrict__ WvT,
                const float* __restrict__ bq, const float* __restrict__ bk,
                const float* __restrict__ bv,
                unsigned short* __restrict__ Qb,   // [B,H,T,Hd], pre-scaled by Hd^-0.5
                unsigned short* __restrict__ Kb,   // [B,H,T,Hd]
                unsigned short* __restrict__ Vt) { // [B,H,Hd,T]  (transposed)
  __shared__ unsigned short wsm[2 * 3 * 64 * 64];  // 48 KB, single shared object
  const int tid  = threadIdx.x;
  const int lane = tid & 31;
  const int wave = tid >> 5;
  const int ln   = lane & 15;
  const int h    = blockIdx.x;                 // 64-col group == head
  const int m0   = blockIdx.y * 128 + wave * 16;
  const int colbase = h * 64;

  const unsigned short* arow = xb + (size_t)(m0 + ln) * D;

  v8f aq[4], ak[4], av[4];
#pragma unroll
  for (int nt = 0; nt < 4; ++nt) { aq[nt] = vzero(); ak[nt] = vzero(); av[nt] = vzero(); }

  // Cooperative async stage of one K=64 slice of WqT/WkT/WvT.
  auto issue = [&](int st, int k0) {
    const int stw = st * 12288;                  // ushort units (24KB per stage)
#pragma unroll
    for (int mat = 0; mat < 3; ++mat) {
      const unsigned short* wp = (mat == 0) ? WqT : ((mat == 1) ? WkT : WvT);
#pragma unroll
      for (int i = 0; i < 2; ++i) {
        int seg = tid + i * 256;                 // 0..511 : 64 rows x 8 x 16B
        int r = seg >> 3, sg = seg & 7;
        const unsigned short* g = wp + (size_t)(colbase + r) * D + k0 + sg * 8;
        async_g2l_b128(&wsm[stw + mat * 4096 + r * 64 + sg * 8], g);
      }
    }
  };

  issue(0, 0);
  int st = 0;
  for (int k0 = 0; k0 < D; k0 += 64, st ^= 1) {
    wait_async0();
    __syncthreads();
    if (k0 + 64 < D) issue(st ^ 1, k0 + 64);

    v16bf a0 = loadA(arow, k0, lane);
    v16bf a1 = loadA(arow, k0 + 32, lane);
    const unsigned short* sb = wsm + (size_t)st * 12288;  // ushort units
#pragma unroll
    for (int nt = 0; nt < 4; ++nt) {
      const unsigned short* rq = sb + (size_t)(nt * 16 + ln) * 64;
      v16bf bq0 = loadB(rq, 0, lane);
      v16bf bq1 = loadB(rq, 32, lane);
      v16bf bk0 = loadB(rq + 4096, 0, lane);
      v16bf bk1 = loadB(rq + 4096, 32, lane);
      v16bf bv0 = loadB(rq + 8192, 0, lane);
      v16bf bv1 = loadB(rq + 8192, 32, lane);
      aq[nt] = wmma_bf16(a0, bq0, aq[nt]);
      aq[nt] = wmma_bf16(a1, bq1, aq[nt]);
      ak[nt] = wmma_bf16(a0, bk0, ak[nt]);
      ak[nt] = wmma_bf16(a1, bk1, ak[nt]);
      av[nt] = wmma_bf16(a0, bv0, av[nt]);
      av[nt] = wmma_bf16(a1, bv1, av[nt]);
    }
    // Pipeline shape: all global loads (async staging + A) up front, then
    // 4-operand DS prolog, then (2 DS reads : 1 WMMA) steady state, WMMA drain.
    SGB(0x020, 10);                 // 6 async-to-LDS + 4 A loads
    SGB(0x100, 8);                  // 4 B operands in flight
    PIPE5 PIPE5 PIPE5 PIPE5         // 20 x (ds2, wmma1)
    SGB(0x008, 4);                  // drain
  }

  const int rbase = m0 + ((lane >= 16) ? 8 : 0);
#pragma unroll
  for (int nt = 0; nt < 4; ++nt) {
    const int hd  = nt * 16 + ln;
    const int col = colbase + hd;
    const float vbq = bq[col], vbk = bk[col], vbv = bv[col];
#pragma unroll
    for (int r = 0; r < 8; ++r) {
      const int row = rbase + r;
      const int bb = row >> 11, t = row & (T - 1);
      const size_t bh = (size_t)(bb * H + h);
      Qb[(bh * T + t) * HD + hd] = f2bf((aq[nt][r] + vbq) * 0.125f); // fold Hd^-0.5
      Kb[(bh * T + t) * HD + hd] = f2bf(ak[nt][r] + vbk);
      Vt[(bh * HD + hd) * T + t] = f2bf(av[nt][r] + vbv);
    }
  }
}

// ---------------------------------------------------------------- flash attention
__global__ __launch_bounds__(256, 1)
void attn_kernel(const unsigned short* __restrict__ Qb,
                 const unsigned short* __restrict__ Kb,
                 const unsigned short* __restrict__ Vt,
                 const unsigned char* __restrict__ mask, // [B,T] bool
                 unsigned short* __restrict__ AO) {      // [B,T,D] bf16
  // [0, 16KB): K/V stages (2 x (4KB K + 4KB V)); [16KB, 24KB): per-wave P tiles.
  __shared__ unsigned short smem[12288];
  const int tid  = threadIdx.x;
  const int lane = tid & 31;
  const int wave = tid >> 5;
  const int ln   = lane & 15;
  const int bh   = blockIdx.y;
  const int bb   = bh >> 4;
  const int h    = bh & 15;
  const int q0   = blockIdx.x * 128 + wave * 16;

  const unsigned short* qrow = Qb + ((size_t)bh * T + (q0 + ln)) * HD;
  const v16bf qa0 = loadA(qrow, 0, lane);
  const v16bf qa1 = loadA(qrow, 32, lane);

  v8f acc[4];
#pragma unroll
  for (int j = 0; j < 4; ++j) acc[j] = vzero();
  float mrow[8], lrow[8];
#pragma unroll
  for (int r = 0; r < 8; ++r) { mrow[r] = -3.0e38f; lrow[r] = 0.f; }

  const unsigned char* mrowp = mask + (size_t)bb * T;
  unsigned short* myp = smem + 8192 + wave * 512;  // 16x32 bf16 P tile

  // Stage one 32-key chunk: K rows [tk..tk+31] x Hd and Vt rows [Hd] x [tk..tk+31].
  auto issueKV = [&](int st, int tk) {
    const int stw = st * 4096;                   // ushort units (8KB per stage)
    {
      int r = tid >> 3, sg = tid & 7;            // 32 rows x 8 x 16B
      const unsigned short* g = Kb + ((size_t)bh * T + tk + r) * HD + sg * 8;
      async_g2l_b128(&smem[stw + r * 64 + sg * 8], g);
    }
    {
      int r = tid >> 2, sg = tid & 3;            // 64 rows x 4 x 16B
      const unsigned short* g = Vt + ((size_t)bh * HD + r) * T + tk + sg * 8;
      async_g2l_b128(&smem[stw + 2048 + r * 32 + sg * 8], g);
    }
  };

  issueKV(0, 0);
  int st = 0;
  for (int tk = 0; tk < T; tk += 32, st ^= 1) {
    wait_async0();
    __syncthreads();
    if (tk + 32 < T) issueKV(st ^ 1, tk + 32);

    const unsigned short* kvK = smem + (size_t)st * 4096;  // ushort units
    const unsigned short* kvV = kvK + 2048;

    // S = Q @ K^T (16x32, two 16-col tiles).
    v8f s0 = vzero(), s1 = vzero();
    {
      const unsigned short* r0 = kvK + (size_t)ln * 64;
      const unsigned short* r1 = kvK + (size_t)(16 + ln) * 64;
      v16bf b00 = loadB(r0, 0, lane);
      v16bf b01 = loadB(r0, 32, lane);
      v16bf b10 = loadB(r1, 0, lane);
      v16bf b11 = loadB(r1, 32, lane);
      s0 = wmma_bf16(qa0, b00, s0);
      s0 = wmma_bf16(qa1, b01, s0);
      s1 = wmma_bf16(qa0, b10, s1);
      s1 = wmma_bf16(qa1, b11, s1);
    }

    // Prefetch the 4 V operands now; their LDS latency hides under softmax VALU.
    v16bf vb[4];
#pragma unroll
    for (int j = 0; j < 4; ++j)
      vb[j] = loadB(kvV + (size_t)(j * 16 + ln) * 32, 0, lane);

    const float pen0 = mrowp[tk + ln]      ? -1.0e9f : 0.f;
    const float pen1 = mrowp[tk + 16 + ln] ? -1.0e9f : 0.f;
#pragma unroll
    for (int r = 0; r < 8; ++r) { s0[r] += pen0; s1[r] += pen1; }

    // Online softmax: rows live across 16-lane halves; reduce with shfl_xor 1..8.
    float p0[8], p1[8];
#pragma unroll
    for (int r = 0; r < 8; ++r) {
      float mx = fmaxf(s0[r], s1[r]);
#pragma unroll
      for (int d = 1; d < 16; d <<= 1) mx = fmaxf(mx, __shfl_xor(mx, d, 32));
      const float mnew  = fmaxf(mrow[r], mx);
      const float alpha = __expf(mrow[r] - mnew);
      p0[r] = __expf(s0[r] - mnew);
      p1[r] = __expf(s1[r] - mnew);
      float rs = p0[r] + p1[r];
#pragma unroll
      for (int d = 1; d < 16; d <<= 1) rs += __shfl_xor(rs, d, 32);
      lrow[r] = lrow[r] * alpha + rs;
      mrow[r] = mnew;
#pragma unroll
      for (int j = 0; j < 4; ++j) acc[j][r] *= alpha;
    }

    // Re-layout P: C/D layout -> LDS (row-major 16x32 bf16) -> A-operand layout.
    {
      const int rb = (lane >= 16) ? 8 : 0;
#pragma unroll
      for (int r = 0; r < 8; ++r) {
        myp[(rb + r) * 32 + ln]      = f2bf(p0[r]);
        myp[(rb + r) * 32 + 16 + ln] = f2bf(p1[r]);
      }
    }
    __asm__ volatile("s_wait_dscnt 0x0" ::: "memory");  // same-wave LDS RAW
    const v16bf pa = loadA(myp + ln * 32, 0, lane);

    // O += P @ V
#pragma unroll
    for (int j = 0; j < 4; ++j)
      acc[j] = wmma_bf16(pa, vb[j], acc[j]);

    // Body pipeline: vmem first, S ds + S wmma, V ds prefetch, P writes,
    // Pa reads, PV wmma.  (VALU floats freely between groups.)
    SGB(0x020, 4);    // 2 async-to-LDS + 2 mask loads
    SGB(0x100, 8);    // S operands
    SGB(0x008, 4);    // S WMMAs
    SGB(0x100, 8);    // V operands (prefetch under softmax)
    SGB(0x200, 16);   // P tile stores
    SGB(0x100, 2);    // Pa operand
    SGB(0x008, 4);    // PV WMMAs
  }

  const int rb = (lane >= 16) ? 8 : 0;
#pragma unroll
  for (int j = 0; j < 4; ++j) {
    const int col = h * 64 + j * 16 + ln;
#pragma unroll
    for (int r = 0; r < 8; ++r) {
      const int t = q0 + rb + r;
      AO[((size_t)bb * T + t) * D + col] = f2bf(acc[j][r] / lrow[r]);
    }
  }
}

// ---------------------------------------------------------------- output projection
__global__ __launch_bounds__(256, 1)
void oproj_kernel(const unsigned short* __restrict__ AO,
                  const unsigned short* __restrict__ WoT,
                  const float* __restrict__ bo,
                  float* __restrict__ out) {
  __shared__ unsigned short wsm[2 * 64 * 64];  // 16 KB: 2 stages of 64 cols x 64 K
  const int tid  = threadIdx.x;
  const int lane = tid & 31;
  const int wave = tid >> 5;
  const int ln   = lane & 15;
  const int nb   = blockIdx.x;
  const int m0   = blockIdx.y * 128 + wave * 16;
  const int colbase = nb * 64;

  const unsigned short* arow = AO + (size_t)(m0 + ln) * D;
  v8f acc[4];
#pragma unroll
  for (int nt = 0; nt < 4; ++nt) acc[nt] = vzero();

  auto issue = [&](int st, int k0) {
    const int stw = st * 4096;                   // ushort units (8KB per stage)
#pragma unroll
    for (int i = 0; i < 2; ++i) {
      int seg = tid + i * 256;                   // 0..511
      int r = seg >> 3, sg = seg & 7;
      const unsigned short* g = WoT + (size_t)(colbase + r) * D + k0 + sg * 8;
      async_g2l_b128(&wsm[stw + r * 64 + sg * 8], g);
    }
  };

  issue(0, 0);
  int st = 0;
  for (int k0 = 0; k0 < D; k0 += 64, st ^= 1) {
    wait_async0();
    __syncthreads();
    if (k0 + 64 < D) issue(st ^ 1, k0 + 64);

    v16bf a0 = loadA(arow, k0, lane);
    v16bf a1 = loadA(arow, k0 + 32, lane);
    const unsigned short* sb = wsm + (size_t)st * 4096;
#pragma unroll
    for (int np = 0; np < 2; ++np) {             // 2 nt at a time: 4 loads, 4 WMMAs
      const unsigned short* r0 = sb + (size_t)((np * 2 + 0) * 16 + ln) * 64;
      const unsigned short* r1 = sb + (size_t)((np * 2 + 1) * 16 + ln) * 64;
      v16bf b00 = loadB(r0, 0, lane);
      v16bf b01 = loadB(r0, 32, lane);
      v16bf b10 = loadB(r1, 0, lane);
      v16bf b11 = loadB(r1, 32, lane);
      acc[np * 2 + 0] = wmma_bf16(a0, b00, acc[np * 2 + 0]);
      acc[np * 2 + 0] = wmma_bf16(a1, b01, acc[np * 2 + 0]);
      acc[np * 2 + 1] = wmma_bf16(a0, b10, acc[np * 2 + 1]);
      acc[np * 2 + 1] = wmma_bf16(a1, b11, acc[np * 2 + 1]);
    }
    SGB(0x020, 6);                  // 2 async-to-LDS + 4 A loads
    SGB(0x100, 4);                  // 2 B operands in flight
    PIPE5 PIPE1                     // 6 x (ds2, wmma1)
    SGB(0x008, 2);                  // drain
  }

  const int rbase = m0 + ((lane >= 16) ? 8 : 0);
#pragma unroll
  for (int nt = 0; nt < 4; ++nt) {
    const int col = colbase + nt * 16 + ln;
    const float vb = bo[col];
#pragma unroll
    for (int r = 0; r < 8; ++r)
      out[(size_t)(rbase + r) * D + col] = acc[nt][r] + vb;
  }
}

// ---------------------------------------------------------------- launch
extern "C" void kernel_launch(void* const* d_in, const int* in_sizes, int n_in,
                              void* d_out, int out_size, void* d_ws, size_t ws_size,
                              hipStream_t stream) {
  const float*         x    = (const float*)d_in[0];
  const unsigned char* mask = (const unsigned char*)d_in[1];
  const float* Wq = (const float*)d_in[2];
  const float* bq = (const float*)d_in[3];
  const float* Wk = (const float*)d_in[4];
  const float* bk = (const float*)d_in[5];
  const float* Wv = (const float*)d_in[6];
  const float* bv = (const float*)d_in[7];
  const float* Wo = (const float*)d_in[8];
  const float* bo = (const float*)d_in[9];
  float* out = (float*)d_out;

  char* ws = (char*)d_ws;
  size_t o = 0;
  unsigned short* xb  = (unsigned short*)(ws + o); o += (size_t)BT * D * 2;  // 16 MB
  unsigned short* WqT = (unsigned short*)(ws + o); o += (size_t)D * D * 2;   //  2 MB
  unsigned short* WkT = (unsigned short*)(ws + o); o += (size_t)D * D * 2;
  unsigned short* WvT = (unsigned short*)(ws + o); o += (size_t)D * D * 2;
  unsigned short* WoT = (unsigned short*)(ws + o); o += (size_t)D * D * 2;
  unsigned short* Qb  = (unsigned short*)(ws + o); o += (size_t)BT * D * 2;  // 16 MB
  unsigned short* Kb  = (unsigned short*)(ws + o); o += (size_t)BT * D * 2;
  unsigned short* Vt  = (unsigned short*)(ws + o); o += (size_t)BT * D * 2;
  unsigned short* AO  = (unsigned short*)(ws + o); o += (size_t)BT * D * 2;  // total 88 MB

  cvt_bf16_kernel<<<(BT * D / 4 + 255) / 256, 256, 0, stream>>>(x, xb, BT * D);
  transpose_bf16_kernel<<<D * D / 256, 256, 0, stream>>>(Wq, WqT);
  transpose_bf16_kernel<<<D * D / 256, 256, 0, stream>>>(Wk, WkT);
  transpose_bf16_kernel<<<D * D / 256, 256, 0, stream>>>(Wv, WvT);
  transpose_bf16_kernel<<<D * D / 256, 256, 0, stream>>>(Wo, WoT);

  qkv_kernel<<<dim3(H, BT / 128), 256, 0, stream>>>(xb, WqT, WkT, WvT, bq, bk, bv,
                                                    Qb, Kb, Vt);
  attn_kernel<<<dim3(T / 128, BATCH * H), 256, 0, stream>>>(Qb, Kb, Vt, mask, AO);
  oproj_kernel<<<dim3(D / 64, BT / 128), 256, 0, stream>>>(AO, WoT, bo, out);
}
